// GraphTransformer_40106404610754
// MI455X (gfx1250) — compile-verified, matrix operand order checked by reading, not run on hardware
//
#include <hip/hip_runtime.h>

#define D_IN    32
#define N_HEADS 4
#define D_HEAD  8

typedef __attribute__((ext_vector_type(16))) _Float16 v16h;
typedef __attribute__((ext_vector_type(8)))  float    v8f;

// K-index for element e of the per-lane v16h in the documented 16-bit A/B
// layout: lanes 0-15 hold K={0..7,16..23}, lanes 16-31 hold K={8..15,24..31},
// two K values (lo/hi half) per VGPR.
__device__ __forceinline__ int wmma_kidx(int e, int g) {
    int p = e >> 1, qh = e & 1;
    return ((p >= 4) ? 16 : 0) + g * 8 + 2 * (p & 3) + qh;
}

// Monotone float->uint mapping so unsigned atomicMax == float max.
__device__ __forceinline__ unsigned fmax_encode(float f) {
    unsigned u = __float_as_uint(f);
    return (u & 0x80000000u) ? ~u : (u | 0x80000000u);
}
__device__ __forceinline__ float fmax_decode(unsigned u) {
    unsigned b = (u & 0x80000000u) ? (u ^ 0x80000000u) : ~u;
    return __uint_as_float(b);
}

// ---------------------------------------------------------------------------
// Zero the accumulators (msg, denom) and the running-max buffer.
// ---------------------------------------------------------------------------
__global__ void init_ws_kernel(float* __restrict__ msg, float* __restrict__ denom,
                               unsigned* __restrict__ mbits, int N) {
    int i = blockIdx.x * blockDim.x + threadIdx.x;
    if (i < N * D_IN) msg[i] = 0.0f;
    if (i < N * N_HEADS) { denom[i] = 0.0f; mbits[i] = 0u; }  // 0 == encoded -min
}

// ---------------------------------------------------------------------------
// QKV projection: one wave computes one 16-row tile of q,k,v via
// v_wmma_f32_16x16x32_f16 (K=32 in a single WMMA; N=32 -> 2 column tiles).
// ---------------------------------------------------------------------------
__global__ void proj_qkv_wmma(const float* __restrict__ x,
                              const float* __restrict__ Wq,
                              const float* __restrict__ Wk,
                              const float* __restrict__ Wv,
                              float* __restrict__ q,
                              float* __restrict__ k,
                              float* __restrict__ v,
                              int nTiles, int N) {
    const int wave = threadIdx.x >> 5;
    const int lane = threadIdx.x & 31;
    const int tile = blockIdx.x * (blockDim.x >> 5) + wave;
    if (tile >= nTiles) return;                 // wave-uniform: EXEC stays all-1
    const int row0 = tile * 16;
    const int m = lane & 15;
    const int g = lane >> 4;

    // A tile: rows of x, f32 -> f16
    const int arow = min(row0 + m, N - 1);
    v16h a;
#pragma unroll
    for (int e = 0; e < 16; ++e)
        a[e] = (_Float16)x[arow * D_IN + wmma_kidx(e, g)];

    const float* W[3]   = {Wq, Wk, Wv};
    float*       dst[3] = {q, k, v};
#pragma unroll
    for (int w = 0; w < 3; ++w) {
#pragma unroll
        for (int t = 0; t < 2; ++t) {
            v16h b;
#pragma unroll
            for (int e = 0; e < 16; ++e)
                b[e] = (_Float16)W[w][wmma_kidx(e, g) * D_IN + t * 16 + m];
            v8f c = {};
            c = __builtin_amdgcn_wmma_f32_16x16x32_f16(
                    false, a, false, b, (short)0, c, false, false);
#pragma unroll
            for (int r = 0; r < 8; ++r) {
                int mo = row0 + r + 8 * g;
                if (mo < N) dst[w][mo * D_IN + t * 16 + m] = c[r];
            }
        }
    }
}

// ---------------------------------------------------------------------------
// Pass 1 over edges: per (edge, head) score -> segment max via uint atomicMax.
// q/k are L2-resident (12.8 MB each), edge list streams from HBM.
// ---------------------------------------------------------------------------
__global__ void edge_scores_max(const long long* __restrict__ edge,
                                const float* __restrict__ q,
                                const float* __restrict__ k,
                                unsigned* __restrict__ mbits,
                                long long E) {
    long long i = (long long)blockIdx.x * blockDim.x + threadIdx.x;
    if (i >= E * N_HEADS) return;
    long long e = i >> 2;
    int h = (int)(i & 3);
    int s = (int)edge[e];
    int d = (int)edge[E + e];
    const float4* qr = (const float4*)(q + (size_t)d * D_IN + h * D_HEAD);
    const float4* kr = (const float4*)(k + (size_t)s * D_IN + h * D_HEAD);
    float4 q0 = qr[0], q1 = qr[1], k0 = kr[0], k1 = kr[1];
    float dot = q0.x * k0.x + q0.y * k0.y + q0.z * k0.z + q0.w * k0.w
              + q1.x * k1.x + q1.y * k1.y + q1.z * k1.z + q1.w * k1.w;
    float score = dot * 0.35355339059327373f;   // 1/sqrt(D_HEAD)
    atomicMax(&mbits[(size_t)d * N_HEADS + h], fmax_encode(score));
}

// ---------------------------------------------------------------------------
// Pass 2 over edges: recompute score (q/k hit L2), exp, scatter-add denom and
// weighted v into msg.
// ---------------------------------------------------------------------------
__global__ void edge_exp_accum(const long long* __restrict__ edge,
                               const float* __restrict__ q,
                               const float* __restrict__ k,
                               const float* __restrict__ v,
                               const unsigned* __restrict__ mbits,
                               float* __restrict__ denom,
                               float* __restrict__ msg,
                               long long E) {
    long long i = (long long)blockIdx.x * blockDim.x + threadIdx.x;
    if (i >= E * N_HEADS) return;
    long long e = i >> 2;
    int h = (int)(i & 3);
    int s = (int)edge[e];
    int d = (int)edge[E + e];
    const float4* qr = (const float4*)(q + (size_t)d * D_IN + h * D_HEAD);
    const float4* kr = (const float4*)(k + (size_t)s * D_IN + h * D_HEAD);
    float4 q0 = qr[0], q1 = qr[1], k0 = kr[0], k1 = kr[1];
    float dot = q0.x * k0.x + q0.y * k0.y + q0.z * k0.z + q0.w * k0.w
              + q1.x * k1.x + q1.y * k1.y + q1.z * k1.z + q1.w * k1.w;
    float score = dot * 0.35355339059327373f;
    float mx = fmax_decode(mbits[(size_t)d * N_HEADS + h]);
    float w = __expf(score - mx);
    atomicAdd(&denom[(size_t)d * N_HEADS + h], w);
    const float4* vr = (const float4*)(v + (size_t)s * D_IN + h * D_HEAD);
    float4 v0 = vr[0], v1 = vr[1];
    float* mr = msg + (size_t)d * D_IN + h * D_HEAD;
    atomicAdd(&mr[0], w * v0.x); atomicAdd(&mr[1], w * v0.y);
    atomicAdd(&mr[2], w * v0.z); atomicAdd(&mr[3], w * v0.w);
    atomicAdd(&mr[4], w * v1.x); atomicAdd(&mr[5], w * v1.y);
    atomicAdd(&mr[6], w * v1.z); atomicAdd(&mr[7], w * v1.w);
}

// ---------------------------------------------------------------------------
// Finalize: attn = msg/denom, out = x + attn @ Wo, via WMMA per 16-row tile.
// Nodes with no in-edges: denom==0 and msg==0 -> attn==0 -> out==x (matches
// the reference semantics).
// ---------------------------------------------------------------------------
__global__ void finalize_wmma(const float* __restrict__ x,
                              const float* __restrict__ msg,
                              const float* __restrict__ denom,
                              const float* __restrict__ Wo,
                              float* __restrict__ out,
                              int nTiles, int N) {
    const int wave = threadIdx.x >> 5;
    const int lane = threadIdx.x & 31;
    const int tile = blockIdx.x * (blockDim.x >> 5) + wave;
    if (tile >= nTiles) return;
    const int row0 = tile * 16;
    const int m = lane & 15;
    const int g = lane >> 4;

    const int arow = min(row0 + m, N - 1);
    v16h a;
#pragma unroll
    for (int e = 0; e < 16; ++e) {
        int kk = wmma_kidx(e, g);
        float den = denom[arow * N_HEADS + (kk >> 3)];
        float val = msg[arow * D_IN + kk] / fmaxf(den, 1e-16f);
        a[e] = (_Float16)val;
    }

#pragma unroll
    for (int t = 0; t < 2; ++t) {
        v16h b;
#pragma unroll
        for (int e = 0; e < 16; ++e)
            b[e] = (_Float16)Wo[wmma_kidx(e, g) * D_IN + t * 16 + m];
        v8f c = {};
        c = __builtin_amdgcn_wmma_f32_16x16x32_f16(
                false, a, false, b, (short)0, c, false, false);
#pragma unroll
        for (int r = 0; r < 8; ++r) {
            int mo = row0 + r + 8 * g;
            if (mo < N)
                out[mo * D_IN + t * 16 + m] = x[mo * D_IN + t * 16 + m] + c[r];
        }
    }
}

// ---------------------------------------------------------------------------
extern "C" void kernel_launch(void* const* d_in, const int* in_sizes, int n_in,
                              void* d_out, int out_size, void* d_ws, size_t ws_size,
                              hipStream_t stream) {
    const float*     x    = (const float*)d_in[0];
    const float*     Wq   = (const float*)d_in[1];
    const float*     Wk   = (const float*)d_in[2];
    const float*     Wv   = (const float*)d_in[3];
    const float*     Wo   = (const float*)d_in[4];
    const long long* edge = (const long long*)d_in[5];

    const int       N = in_sizes[0] / D_IN;
    const long long E = (long long)in_sizes[5] / 2;

    // Workspace layout (all L2-resident: ~54 MB total vs 192 MB L2)
    float*    q     = (float*)d_ws;
    float*    k     = q   + (size_t)N * D_IN;
    float*    v     = k   + (size_t)N * D_IN;
    float*    msg   = v   + (size_t)N * D_IN;
    float*    denom = msg + (size_t)N * D_IN;
    unsigned* mbits = (unsigned*)(denom + (size_t)N * N_HEADS);

    const int nTiles = (N + 15) / 16;

    init_ws_kernel<<<(N * D_IN + 255) / 256, 256, 0, stream>>>(msg, denom, mbits, N);

    proj_qkv_wmma<<<(nTiles + 3) / 4, 128, 0, stream>>>(x, Wq, Wk, Wv, q, k, v, nTiles, N);

    const long long work = E * N_HEADS;
    const int eblocks = (int)((work + 255) / 256);
    edge_scores_max<<<eblocks, 256, 0, stream>>>(edge, q, k, mbits, E);
    edge_exp_accum<<<eblocks, 256, 0, stream>>>(edge, q, k, v, mbits, denom, msg, E);

    finalize_wmma<<<(nTiles + 3) / 4, 128, 0, stream>>>(x, msg, denom, Wo,
                                                        (float*)d_out, nTiles, N);
}
